// RNNModel_53678501265709
// MI455X (gfx1250) — matrix operand order repaired and checked by reading, not compiled, not run on hardware
//
#include <hip/hip_runtime.h>

#define Hh 256          // GRU units
#define Tt 512          // timesteps (enc and dec)
#define Bb 256          // batch
#define NC 768          // 3*H gate columns
#define BT (Bb*Tt)      // 131072 flattened rows

typedef __attribute__((ext_vector_type(16))) __bf16 v16bf;
typedef __attribute__((ext_vector_type(8)))  __bf16 v8bf;
typedef __attribute__((ext_vector_type(8)))  float  v8f;

#define VZ8 {0.f,0.f,0.f,0.f,0.f,0.f,0.f,0.f}

__device__ __forceinline__ v16bf cat16(v8bf lo, v8bf hi) {
  return __builtin_shufflevector(lo, hi, 0,1,2,3,4,5,6,7,8,9,10,11,12,13,14,15);
}
// B fragment: lane holds K = [k0..k0+7] and [k0+16..k0+23] (K-major rows of permuted W/U)
__device__ __forceinline__ v16bf ldB(const __bf16* p) {
  v8bf lo = *(const v8bf*)p;
  v8bf hi = *(const v8bf*)(p + 16);
  return cat16(lo, hi);
}
// A fragment: lane holds K = [k0..k0+7] and [k0+16..k0+23] (ISA 16-bit A layout)
__device__ __forceinline__ v16bf ldA(const __bf16* p) {
  v8bf lo = *(const v8bf*)p;
  v8bf hi = *(const v8bf*)(p + 16);
  return cat16(lo, hi);
}
__device__ __forceinline__ v8f wmma_bf(v16bf a, v16bf b, v8f c) {
  return __builtin_amdgcn_wmma_f32_16x16x32_bf16(false, a, false, b, (short)0, c,
                                                 false, false);
}
__device__ __forceinline__ float sigm(float x) { return 1.0f / (1.0f + __expf(-x)); }

// CDNA5 async DMA: global -> LDS, 16B per lane, tracked by ASYNCcnt
__device__ __forceinline__ void async_ld16(unsigned int lds_off, const void* g) {
  asm volatile("global_load_async_to_lds_b128 %0, %1, off"
               :: "v"(lds_off), "v"((unsigned long long)(size_t)g)
               : "memory");
}
__device__ __forceinline__ void wait_async_le3() {
#if __has_builtin(__builtin_amdgcn_s_wait_asynccnt)
  __builtin_amdgcn_s_wait_asynccnt(3);
#else
  asm volatile("s_wait_asynccnt 0x3" ::: "memory");
#endif
}

// ---- weight prep: [K=256][768] f32 -> [768][256] bf16 with gate-tile column permute
// permuted col nc = tile*48 + gate*16 + j  <->  original col = gate*256 + tile*16 + j
__global__ void conv_wu(const float* __restrict__ src, const float* __restrict__ bsrc,
                        __bf16* __restrict__ dst, float* __restrict__ bdst) {
  int nc = blockIdx.x;            // 0..767
  int k  = threadIdx.x;           // 0..255
  int tile = nc / 48, rem = nc % 48, g = rem / 16, j = rem % 16;
  int oc = g * Hh + tile * 16 + j;
  dst[(size_t)nc * Hh + k] = (__bf16)src[(size_t)k * NC + oc];
  if (k == 0) bdst[nc] = bsrc[oc];
}

// ---- xg for tiny K (11 or 5): plain FMA, writes permuted bf16 [t][b][nc]
__global__ void xg_small(const float* __restrict__ x, const float* __restrict__ Wf,
                         const float* __restrict__ bi, __bf16* __restrict__ xg, int K) {
  size_t idx = (size_t)blockIdx.x * 256 + threadIdx.x;
  if (idx >= (size_t)BT * NC) return;
  int nc = (int)(idx % NC);
  size_t m = idx / NC;
  int b = (int)(m % Bb), t = (int)(m / Bb);
  int tile = nc / 48, rem = nc % 48, g = rem / 16, j = rem % 16;
  int oc = g * Hh + tile * 16 + j;
  const float* xr = x + ((size_t)b * Tt + t) * K;
  float s = bi[oc];
  for (int d = 0; d < K; ++d) s += xr[d] * Wf[(size_t)d * NC + oc];
  xg[idx] = (__bf16)s;
}

// ---- xg for K=256 layers: WMMA GEMM. A rows are [b*T+t][256] bf16.
__global__ __launch_bounds__(512)
void xg_gemm(const __bf16* __restrict__ A, const __bf16* __restrict__ Wp,
             const float* __restrict__ bip, __bf16* __restrict__ xg) {
  int tid = threadIdx.x, w = tid >> 5, lane = tid & 31;
  int mbase = blockIdx.x * 16;
  int n = lane & 15;
  int kh16 = (lane & 16) ? 16 : 0;
  int kh8  = (lane & 16) ? 8  : 0;
  const __bf16* wz = Wp + (size_t)(w * 48 +      n) * Hh;
  const __bf16* wr = Wp + (size_t)(w * 48 + 16 + n) * Hh;
  const __bf16* wh = Wp + (size_t)(w * 48 + 32 + n) * Hh;
  const __bf16* ap = A + (size_t)(mbase + (lane & 15)) * Hh;
  v8f az = VZ8, arr = VZ8, ah = VZ8;
#pragma unroll
  for (int kb = 0; kb < 8; ++kb) {
    v16bf a = ldA(ap + kb * 32 + kh8);
    az  = wmma_bf(a, ldB(wz + kb * 32 + kh8), az);
    arr = wmma_bf(a, ldB(wr + kb * 32 + kh8), arr);
    ah  = wmma_bf(a, ldB(wh + kb * 32 + kh8), ah);
  }
  float bz = bip[w * 48 + n], br = bip[w * 48 + 16 + n], bhv = bip[w * 48 + 32 + n];
  int mrow = (lane >> 4) << 3;
#pragma unroll
  for (int rr = 0; rr < 8; ++rr) {
    int m = mbase + mrow + rr;
    int b = m >> 9, t = m & 511;               // m = b*T + t
    __bf16* o = xg + ((size_t)t * Bb + b) * NC + w * 48;
    o[n]      = (__bf16)(az[rr]  + bz);
    o[16 + n] = (__bf16)(arr[rr] + br);
    o[32 + n] = (__bf16)(ah[rr]  + bhv);
  }
}

// ---- recurrent scan: 16 WGs x 16 batch rows; 16 waves each own 16 hidden units.
// f32 master h-state lives in registers (each (row,hidden) cell owned by 1 lane);
// bf16 mirror in LDS feeds WMMA A-fragments; xg staged by async DMA, double-buffered.
#define SB (Hh + 8)                 // bf16 h-state LDS stride (bank-conflict pad)
#define XTILE_BYTES (16 * NC * 2)   // 24576 per buffer

__global__ __launch_bounds__(512)
void gru_scan(const __bf16* __restrict__ xg, const __bf16* __restrict__ Up,
              const float* __restrict__ bhp, const float* __restrict__ h_init,
              __bf16* __restrict__ y_out, float* __restrict__ h_final) {
  __shared__ __bf16 hsb[16 * SB];
  __shared__ __align__(16) unsigned char xls[2][XTILE_BYTES];
  int tid = threadIdx.x, w = tid >> 5, lane = tid & 31;
  int bbase = blockIdx.x * 16;
  int n = lane & 15;
  int kh16 = (lane & 16) ? 16 : 0;
  int kh8  = (lane & 16) ? 8  : 0;
  int mrow = (lane >> 4) << 3;
  int hcol = w * 16 + n;
  int arow = lane & 15;

  // init h: f32 master in registers, bf16 mirror in LDS (ownership partitions tile)
  float hreg[8];
#pragma unroll
  for (int rr = 0; rr < 8; ++rr) {
    int row = mrow + rr;
    float v = h_init ? h_init[(size_t)(bbase + row) * Hh + hcol] : 0.0f;
    hreg[rr] = v;
    hsb[row * SB + hcol] = (__bf16)v;
  }

  // U fragments resident in VGPRs for the whole scan (zero weight traffic/step)
  v16bf Bz[8], Br[8], Bh[8];
  const __bf16* uz = Up + (size_t)(w * 48 +      n) * Hh;
  const __bf16* ur = Up + (size_t)(w * 48 + 16 + n) * Hh;
  const __bf16* uh = Up + (size_t)(w * 48 + 32 + n) * Hh;
#pragma unroll
  for (int kb = 0; kb < 8; ++kb) {
    int ko = kb * 32 + kh8;
    Bz[kb] = ldB(uz + ko); Br[kb] = ldB(ur + ko); Bh[kb] = ldB(uh + ko);
  }
  float bz = bhp[w * 48 + n], br = bhp[w * 48 + 16 + n], bh = bhp[w * 48 + 32 + n];

  // stage xg tile for step t into buffer buf (48 B per thread, 3 async b128 ops)
  auto stage = [&](int t, int buf) {
    const unsigned char* g =
        (const unsigned char*)(xg + ((size_t)t * Bb + bbase) * NC) + tid * 48;
    unsigned int l = (unsigned int)(size_t)(&xls[buf][tid * 48]);
    async_ld16(l,      g);
    async_ld16(l + 16, g + 16);
    async_ld16(l + 32, g + 32);
  };

  stage(0, 0);
  __syncthreads();

  for (int t = 0; t < Tt; ++t) {
    int cur = t & 1;
    int tn = (t + 1 < Tt) ? t + 1 : t;   // clamped: keeps 6-outstanding invariant
    stage(tn, cur ^ 1);                  // overlap next-step DMA with WMMA below
    v8f az = VZ8, arr = VZ8, ah = VZ8;
#pragma unroll
    for (int kb = 0; kb < 8; ++kb) {
      v16bf a = ldA(hsb + arow * SB + kb * 32 + kh8);
      az  = wmma_bf(a, Bz[kb], az);
      arr = wmma_bf(a, Br[kb], arr);
      ah  = wmma_bf(a, Bh[kb], ah);
    }
    wait_async_le3();   // own stage(t) DMA complete (in-order), stage(t+1) may fly
    __syncthreads();    // all waves: h(t) reads done AND stage(t) visible in LDS
    const __bf16* xt = (const __bf16*)(xls[cur]);
#pragma unroll
    for (int rr = 0; rr < 8; ++rr) {
      int row = mrow + rr;
      const __bf16* xr = xt + row * NC + w * 48;
      float xz = (float)xr[n], xrr = (float)xr[16 + n], xh = (float)xr[32 + n];
      float hprev = hreg[rr];
      float z  = sigm(az[rr]  + bz + xz);
      float rg = sigm(arr[rr] + br + xrr);
      float cand = fmaxf(0.0f, xh + rg * (ah[rr] + bh));
      float hn = z * hprev + (1.0f - z) * cand;
      hreg[rr] = hn;
      hsb[row * SB + hcol] = (__bf16)hn;
      if (y_out) y_out[((size_t)(bbase + row) * Tt + t) * Hh + hcol] = (__bf16)hn;
    }
    __syncthreads();    // h(t+1) visible; also fences xls[cur] reuse next step
  }
  if (h_final) {
#pragma unroll
    for (int rr = 0; rr < 8; ++rr)
      h_final[(size_t)(bbase + mrow + rr) * Hh + hcol] = hreg[rr];
  }
}

// ---- dense head: [BT][256] bf16 @ [256][5] + b -> f32
__global__ void dense_k(const __bf16* __restrict__ d1, const float* __restrict__ Wd,
                        const float* __restrict__ bd, float* __restrict__ out) {
  size_t idx = (size_t)blockIdx.x * 256 + threadIdx.x;
  if (idx >= (size_t)BT * 5) return;
  int f = (int)(idx % 5);
  size_t m = idx / 5;
  const __bf16* row = d1 + m * Hh;
  float s = bd[f];
  for (int h = 0; h < Hh; ++h) s += (float)row[h] * Wd[(size_t)h * 5 + f];
  out[idx] = s;
}

extern "C" void kernel_launch(void* const* d_in, const int* in_sizes, int n_in,
                              void* d_out, int out_size, void* d_ws, size_t ws_size,
                              hipStream_t stream) {
  (void)in_sizes; (void)n_in; (void)out_size; (void)ws_size;
  const float* enc_x = (const float*)d_in[0];
  const float* dec_x = (const float*)d_in[1];
  const float *Wm[4], *Um[4], *bim[4], *bhm[4];
  for (int l = 0; l < 4; ++l) {
    Wm[l]  = (const float*)d_in[2 + 4 * l];
    Um[l]  = (const float*)d_in[3 + 4 * l];
    bim[l] = (const float*)d_in[4 + 4 * l];
    bhm[l] = (const float*)d_in[5 + 4 * l];
  }
  const float* dW = (const float*)d_in[18];
  const float* db = (const float*)d_in[19];
  float* out = (float*)d_out;

  char* base = (char*)d_ws;
  size_t off = 0;
  auto carve = [&](size_t bytes) -> char* {
    char* p = base + off;
    off += (bytes + 255) & ~(size_t)255;
    return p;
  };
  __bf16* Up[4]; float* bhp[4];
  for (int l = 0; l < 4; ++l) {
    Up[l]  = (__bf16*)carve((size_t)NC * Hh * 2);
    bhp[l] = (float*)carve((size_t)NC * 4);
  }
  __bf16* Wp1 = (__bf16*)carve((size_t)NC * Hh * 2);
  float*  bp1 = (float*)carve((size_t)NC * 4);
  __bf16* Wp3 = (__bf16*)carve((size_t)NC * Hh * 2);
  float*  bp3 = (float*)carve((size_t)NC * 4);
  __bf16* xg  = (__bf16*)carve((size_t)BT * NC * 2);   // 201 MB
  __bf16* yA  = (__bf16*)carve((size_t)BT * Hh * 2);   // 67 MB
  float*  st  = (float*)carve((size_t)Bb * Hh * 4);

  // weight prep (permute + bf16)
  for (int l = 0; l < 4; ++l)
    conv_wu<<<NC, 256, 0, stream>>>(Um[l], bhm[l], Up[l], bhp[l]);
  conv_wu<<<NC, 256, 0, stream>>>(Wm[1], bim[1], Wp1, bp1);
  conv_wu<<<NC, 256, 0, stream>>>(Wm[3], bim[3], Wp3, bp3);

  const int smallBlocks = (int)(((size_t)BT * NC + 255) / 256);  // 393216

  // encoder layer 0
  xg_small<<<smallBlocks, 256, 0, stream>>>(enc_x, Wm[0], bim[0], xg, 11);
  gru_scan<<<Bb / 16, 512, 0, stream>>>(xg, Up[0], bhp[0], nullptr, yA, nullptr);
  // encoder layer 1 (only final state kept)
  xg_gemm<<<BT / 16, 512, 0, stream>>>(yA, Wp1, bp1, xg);
  gru_scan<<<Bb / 16, 512, 0, stream>>>(xg, Up[1], bhp[1], nullptr, nullptr, st);
  // decoder layer 0 (init = enc_state)
  xg_small<<<smallBlocks, 256, 0, stream>>>(dec_x, Wm[2], bim[2], xg, 5);
  gru_scan<<<Bb / 16, 512, 0, stream>>>(xg, Up[2], bhp[2], st, yA, nullptr);
  // decoder layer 1 (init = enc_state)
  xg_gemm<<<BT / 16, 512, 0, stream>>>(yA, Wp3, bp3, xg);
  gru_scan<<<Bb / 16, 512, 0, stream>>>(xg, Up[3], bhp[3], st, yA, nullptr);
  // dense head
  dense_k<<<(int)(((size_t)BT * 5 + 255) / 256), 256, 0, stream>>>(yA, dW, db, out);
}
